// DecoderTARDIS_87445534146928
// MI455X (gfx1250) — matrix operand order repaired and checked by reading, not compiled, not run on hardware
//
#include <hip/hip_runtime.h>
#include <hip/hip_bf16.h>
#include <math.h>
#include <stdint.h>

typedef _Float16 f16;
typedef __attribute__((ext_vector_type(8)))  _Float16 v8h;
typedef __attribute__((ext_vector_type(16))) _Float16 v16h;
typedef __attribute__((ext_vector_type(8)))  float    v8f;

// Problem constants (match reference)
constexpr int LSTEPS = 32;
constexpr int BATCH  = 128;
constexpr int IDIM   = 512;
constexpr int HDIM   = 1024;
constexpr int NSLOT  = 512;
constexpr int ADIM   = 128;
constexpr int CDIM   = 128;
constexpr int ACD    = 256;   // ADIM + CDIM
constexpr int LB     = LSTEPS * BATCH;

__device__ __forceinline__ v16h cat16(v8h lo, v8h hi) {
    return __builtin_shufflevector(lo, hi, 0,1,2,3,4,5,6,7,8,9,10,11,12,13,14,15);
}

// Async global->LDS 16B copy (CDNA5 GLOBAL_LOAD_ASYNC_TO_LDS_B128, ASYNCcnt).
// ldsoff: low 32 bits of a generic pointer into __shared__ (== LDS offset).
__device__ __forceinline__ void async_copy_b128(void* lds_dst, const void* gsrc) {
    uint32_t ldsoff = (uint32_t)(uintptr_t)lds_dst;
    uint64_t gaddr  = (uint64_t)(uintptr_t)gsrc;
    asm volatile("global_load_async_to_lds_b128 %0, %1, off"
                 :: "v"(ldsoff), "v"(gaddr) : "memory");
}
__device__ __forceinline__ void wait_asynccnt0() {
    asm volatile("s_wait_asynccnt 0x0" ::: "memory");
}

// ---------------------------------------------------------------------------
// Generic f16 WMMA GEMM: C[M,N] = A[M,K] * B[K,N] (+ Add[M,N] if ADD)
// block = 256 threads (8 waves), block tile 64x64, wave tile 16x32,
// K-step 32 via v_wmma_f32_16x16x32_f16.  M%64==0, N%64==0, K%32==0.
// A tile staged via async global->LDS DMA; B tile transposed via ds stores.
// ---------------------------------------------------------------------------
template<bool ADD>
__global__ __launch_bounds__(256)
void gemm_f16_wmma(const f16* __restrict__ A, const f16* __restrict__ Bm,
                   const float* __restrict__ Add, float* __restrict__ C,
                   int M, int Nn, int K)
{
    __shared__ __align__(16) f16 lA[64 * 32];     // A tile, row stride 32
    __shared__ __align__(16) f16 lBT[64 * 48];    // B tile transposed (NxK), row stride 48

    const int tid  = threadIdx.x;
    const int wave = tid >> 5;
    const int lane = tid & 31;
    const int wr   = wave >> 1;        // 0..3  (16-row band)
    const int wc   = wave & 1;         // 0..1  (32-col band)
    const int half = lane >> 4;        // 0/1
    const int lr   = lane & 15;
    const int mblk = blockIdx.x * 64;
    const int nblk = blockIdx.y * 64;

    v8f c0 = {}; v8f c1 = {};

    for (int kc = 0; kc < K; kc += 32) {
        { // stage A: 64x32, 16 bytes per thread, async DMA straight into LDS
            int row = tid >> 2;
            int cg  = (tid & 3) * 8;
            const f16* src = A + (size_t)(mblk + row) * K + kc + cg;
            if (kc + 32 < K) __builtin_prefetch(src + 32, 0, 0);
            async_copy_b128(&lA[row * 32 + cg], src);
        }
        { // stage B transposed: rows k, cols n -> lBT[n][k]
            int r  = tid >> 3;          // k  0..31
            int cg = (tid & 7) * 8;     // n  0..56
            const f16* src = Bm + (size_t)(kc + r) * Nn + nblk + cg;
            v8h v = *(const v8h*)src;
            #pragma unroll
            for (int j = 0; j < 8; ++j) lBT[(cg + j) * 48 + r] = v[j];
        }
        wait_asynccnt0();      // async A-tile copies complete
        __syncthreads();

        // A fragment: lane<16 -> K{0..7,16..23}; lane>=16 -> K{8..15,24..31}
        int m = wr * 16 + lr;
        v16h af = cat16(*(const v8h*)(&lA[m * 32 + half * 8]),
                        *(const v8h*)(&lA[m * 32 + 16 + half * 8]));
        // B fragments: lane<16 -> K{0..7,16..23} of col n; lane>=16 -> K{8..15,24..31}
        int n0 = wc * 32 + lr;
        v16h bf0 = cat16(*(const v8h*)(&lBT[n0 * 48 + half * 8]),
                         *(const v8h*)(&lBT[n0 * 48 + 16 + half * 8]));
        int n1 = n0 + 16;
        v16h bf1 = cat16(*(const v8h*)(&lBT[n1 * 48 + half * 8]),
                         *(const v8h*)(&lBT[n1 * 48 + 16 + half * 8]));

        c0 = __builtin_amdgcn_wmma_f32_16x16x32_f16(false, af, false, bf0, (short)0, c0, false, false);
        c1 = __builtin_amdgcn_wmma_f32_16x16x32_f16(false, af, false, bf1, (short)0, c1, false, false);
        __syncthreads();
    }

    #pragma unroll
    for (int v = 0; v < 8; ++v) {
        int mrow = mblk + wr * 16 + half * 8 + v;
        int col0 = nblk + wc * 32 + lr;
        size_t i0 = (size_t)mrow * Nn + col0;
        size_t i1 = i0 + 16;
        float o0 = c0[v], o1 = c1[v];
        if (ADD) { o0 += Add[i0]; o1 += Add[i1]; }
        C[i0] = o0; C[i1] = o1;
    }
}

// ---------------------------------------------------------------------------
// Fused logits kernel:
//   hid_r[b,n,:] = tanh(base[b,:] + addr_part[n,:] + val[b,n,:]@Wm2wC)
//   logits[b,n]  = hid_r[b,n,:] . atten
// block = (b, 64-slot chunk); WMMA over K=128, AC swept in 64-col chunks,
// tanh+dot fused into the accumulator epilogue, ds_add_f32 row reduction.
// ---------------------------------------------------------------------------
__global__ __launch_bounds__(256)
void logits_kernel(const float* __restrict__ valmem,   // (B, NSLOT, CDIM)
                   const f16*   __restrict__ WmC,      // (CDIM, ACD)
                   const float* __restrict__ base,     // (B, ACD)
                   const float* __restrict__ addrp,    // (NSLOT, ACD)
                   const float* __restrict__ attn,     // (ACD,)
                   float*       __restrict__ logits)   // (B, NSLOT)
{
    __shared__ __align__(16) f16 lA[64 * 128];     // val tile f16, stride 128
    __shared__ __align__(16) f16 lBT[64 * 136];    // W tile transposed, stride 136
    __shared__ float lsum[64];

    const int tid  = threadIdx.x;
    const int wave = tid >> 5;
    const int lane = tid & 31;
    const int wr   = wave >> 1;
    const int wc   = wave & 1;
    const int half = lane >> 4;
    const int lr   = lane & 15;
    const int b    = blockIdx.x;
    const int nblk = blockIdx.y * 64;

    if (tid < 64) lsum[tid] = 0.0f;

    { // stage val tile once, converting f32 -> f16
        const float* src = valmem + ((size_t)b * NSLOT + nblk) * CDIM;
        for (int i = tid; i < 64 * 128; i += 256) lA[i] = (f16)src[i];
    }

    float ps[8];
    #pragma unroll
    for (int v = 0; v < 8; ++v) ps[v] = 0.0f;

    for (int acc = 0; acc < ACD; acc += 64) {
        __syncthreads();   // covers initial staging + protects lBT reuse
        { // stage W_m2w[A:, acc:acc+64] transposed
            int r  = tid >> 3;
            int cg = (tid & 7) * 8;
            for (int rr = r; rr < 128; rr += 32) {
                v8h v = *(const v8h*)(WmC + (size_t)rr * ACD + acc + cg);
                #pragma unroll
                for (int j = 0; j < 8; ++j) lBT[(cg + j) * 136 + rr] = v[j];
            }
        }
        __syncthreads();

        v8f c0 = {}; v8f c1 = {};
        #pragma unroll
        for (int kc = 0; kc < 128; kc += 32) {
            int m = wr * 16 + lr;
            v16h af = cat16(*(const v8h*)(&lA[m * 128 + kc + half * 8]),
                            *(const v8h*)(&lA[m * 128 + kc + 16 + half * 8]));
            int n0 = wc * 32 + lr;
            v16h bf0 = cat16(*(const v8h*)(&lBT[n0 * 136 + kc + half * 8]),
                             *(const v8h*)(&lBT[n0 * 136 + kc + 16 + half * 8]));
            int n1 = n0 + 16;
            v16h bf1 = cat16(*(const v8h*)(&lBT[n1 * 136 + kc + half * 8]),
                             *(const v8h*)(&lBT[n1 * 136 + kc + 16 + half * 8]));
            c0 = __builtin_amdgcn_wmma_f32_16x16x32_f16(false, af, false, bf0, (short)0, c0, false, false);
            c1 = __builtin_amdgcn_wmma_f32_16x16x32_f16(false, af, false, bf1, (short)0, c1, false, false);
        }

        // fused epilogue: + base + addr_part, tanh, dot with atten
        #pragma unroll
        for (int v = 0; v < 8; ++v) {
            int mloc = wr * 16 + half * 8 + v;
            int nrow = nblk + mloc;
            int col0 = acc + wc * 32 + lr;
            float x0 = c0[v] + base[(size_t)b * ACD + col0]
                             + addrp[(size_t)nrow * ACD + col0];
            float x1 = c1[v] + base[(size_t)b * ACD + col0 + 16]
                             + addrp[(size_t)nrow * ACD + col0 + 16];
            ps[v] += tanhf(x0) * attn[col0] + tanhf(x1) * attn[col0 + 16];
        }
    }

    #pragma unroll
    for (int v = 0; v < 8; ++v) {
        int mloc = wr * 16 + half * 8 + v;
        atomicAdd(&lsum[mloc], ps[v]);
    }
    __syncthreads();
    if (tid < 64) logits[(size_t)b * NSLOT + nblk + tid] = lsum[tid];
}

// ---------------------------------------------------------------------------
// Small helper kernels
// ---------------------------------------------------------------------------
__global__ void f32_to_f16_k(const float* __restrict__ s, f16* __restrict__ d, int n) {
    int i = blockIdx.x * 256 + threadIdx.x;
    if (i < n) d[i] = (f16)s[i];
}
__global__ void zero_f32_k(float* __restrict__ p, int n) {
    int i = blockIdx.x * 256 + threadIdx.x;
    if (i < n) p[i] = 0.0f;
}
__global__ void init_h_k(const float* __restrict__ hid, float* __restrict__ h,
                         f16* __restrict__ hf, int n) {
    int i = blockIdx.x * 256 + threadIdx.x;
    if (i < n) { float v = hid[i]; h[i] = v; hf[i] = (f16)v; }
}
__global__ void extract_addr_k(const float* __restrict__ mb, f16* __restrict__ d) {
    int i = blockIdx.x * 256 + threadIdx.x;
    if (i < NSLOT * ADIM) d[i] = (f16)mb[(i / ADIM) * ACD + (i % ADIM)];
}

// emb @ {W_i2gates (IDIM,3), W_i2ab (IDIM,2)} for all L*B rows
__global__ void emb_small_k(const float* __restrict__ inp,
                            const float* __restrict__ Wg,
                            const float* __restrict__ Wab,
                            float* __restrict__ embG, float* __restrict__ embAB) {
    int i = blockIdx.x * 256 + threadIdx.x;
    if (i >= LB * 5) return;
    int row = i / 5, j = i % 5;
    const float* x = inp + (size_t)row * IDIM;
    float s = 0.0f;
    if (j < 3) {
        for (int k = 0; k < IDIM; ++k) s += x[k] * Wg[k * 3 + j];
        embG[row * 3 + j] = s;
    } else {
        int jj = j - 3;
        for (int k = 0; k < IDIM; ++k) s += x[k] * Wab[k * 2 + jj];
        embAB[row * 2 + jj] = s;
    }
}

// u = layernorm-ish of w_sum over N, emitted directly as f16 for the GEMM
__global__ void unorm_k(const float* __restrict__ w_sum, f16* __restrict__ uh) {
    __shared__ float s1[128], s2[128];
    int b = blockIdx.x, tid = threadIdx.x;
    float a = 0.0f, q = 0.0f;
    for (int n = tid; n < NSLOT; n += 128) {
        float v = w_sum[(size_t)b * NSLOT + n];
        a += v; q += v * v;
    }
    s1[tid] = a; s2[tid] = q;
    __syncthreads();
    for (int s = 64; s > 0; s >>= 1) {
        if (tid < s) { s1[tid] += s1[tid + s]; s2[tid] += s2[tid + s]; }
        __syncthreads();
    }
    float mu  = s1[0] / (float)NSLOT;
    float var = s2[0] / (float)NSLOT - mu * mu;
    float rs  = rsqrtf(var + 1e-5f);
    for (int n = tid; n < NSLOT; n += 128)
        uh[(size_t)b * NSLOT + n] = (f16)((w_sum[(size_t)b * NSLOT + n] - mu) * rs);
}

// argmax over logits row; pick slot, bump w_sum, gather r (addr || val)
__global__ void argmax_pick_k(const float* __restrict__ logits,
                              float* __restrict__ w_sum, int* __restrict__ pos,
                              const float* __restrict__ mem_bias,
                              const float* __restrict__ valmem,
                              float* __restrict__ r, f16* __restrict__ rf16) {
    __shared__ float vmax[128];
    __shared__ int   vidx[128];
    __shared__ int   spos;
    int b = blockIdx.x, tid = threadIdx.x;
    float best = -3.4e38f; int bi = 0;
    for (int n = tid; n < NSLOT; n += 128) {
        float v = logits[(size_t)b * NSLOT + n];
        if (v > best || (v == best && n < bi)) { best = v; bi = n; }
    }
    vmax[tid] = best; vidx[tid] = bi;
    __syncthreads();
    for (int s = 64; s > 0; s >>= 1) {
        if (tid < s) {
            if (vmax[tid + s] > vmax[tid] ||
                (vmax[tid + s] == vmax[tid] && vidx[tid + s] < vidx[tid])) {
                vmax[tid] = vmax[tid + s]; vidx[tid] = vidx[tid + s];
            }
        }
        __syncthreads();
    }
    if (tid == 0) {
        spos = vidx[0];
        pos[b] = spos;
        w_sum[(size_t)b * NSLOT + spos] += 1.0f;
    }
    __syncthreads();
    int pp = spos;
    for (int j = tid; j < ACD; j += 128) {
        float v = (j < ADIM) ? mem_bias[(size_t)pp * ACD + j]
                             : valmem[((size_t)b * NSLOT + pp) * CDIM + (j - ADIM)];
        r[(size_t)b * ACD + j] = v;
        rf16[(size_t)b * ACD + j] = (f16)v;
    }
}

// gates (f,i,o) and hard alpha/beta (ST-gumbel-sigmoid forward value = x>0)
__global__ void gates_k(const float* __restrict__ h, const float* __restrict__ r,
                        const float* __restrict__ Wh2g, const float* __restrict__ Wr2g,
                        const float* __restrict__ Wh2ab, const float* __restrict__ Wr2ab,
                        const float* __restrict__ embG_t, const float* __restrict__ embAB_t,
                        float* __restrict__ fio, float* __restrict__ albe) {
    __shared__ float red[256];
    int b = blockIdx.x, tid = threadIdx.x;
    float p[5] = {0, 0, 0, 0, 0};
    for (int k = tid; k < HDIM; k += 256) {
        float hv = h[(size_t)b * HDIM + k];
        p[0] += hv * Wh2g[k * 3 + 0];
        p[1] += hv * Wh2g[k * 3 + 1];
        p[2] += hv * Wh2g[k * 3 + 2];
        p[3] += hv * Wh2ab[k * 2 + 0];
        p[4] += hv * Wh2ab[k * 2 + 1];
    }
    {   // ACD == 256 == blockDim, one element each
        float rv = r[(size_t)b * ACD + tid];
        p[0] += rv * Wr2g[tid * 3 + 0];
        p[1] += rv * Wr2g[tid * 3 + 1];
        p[2] += rv * Wr2g[tid * 3 + 2];
        p[3] += rv * Wr2ab[tid * 2 + 0];
        p[4] += rv * Wr2ab[tid * 2 + 1];
    }
    for (int j = 0; j < 5; ++j) {
        red[tid] = p[j];
        __syncthreads();
        for (int s = 128; s > 0; s >>= 1) {
            if (tid < s) red[tid] += red[tid + s];
            __syncthreads();
        }
        if (tid == 0) {
            float x = red[0];
            if (j < 3) {
                x += embG_t[b * 3 + j];
                fio[b * 3 + j] = 1.0f / (1.0f + expf(-x));
            } else {
                x += embAB_t[b * 2 + (j - 3)];
                albe[b * 2 + (j - 3)] = (x > 0.0f) ? 1.0f : 0.0f;  // alpha (j==3), beta (j==4)
            }
        }
        __syncthreads();
    }
}

// c_tilde / cc / h update; also writes out[t] and refreshes f16 h
__global__ void state_update_k(const float* __restrict__ P1, const float* __restrict__ P2,
                               const float* __restrict__ embC_t,
                               const float* __restrict__ fio, const float* __restrict__ albe,
                               float* __restrict__ cc, float* __restrict__ h,
                               f16* __restrict__ hf16, float* __restrict__ out_t) {
    int idx = blockIdx.x * 256 + threadIdx.x;    // B*HDIM
    int b = idx >> 10;
    float fg = fio[b * 3 + 0], ig = fio[b * 3 + 1], og = fio[b * 3 + 2];
    float al = albe[b * 2 + 0], be = albe[b * 2 + 1];
    float ct = tanhf(be * P1[idx] + embC_t[idx] + al * P2[idx]);
    float c  = fg * cc[idx] + ig * ct;
    cc[idx]  = c;
    float hh = og * tanhf(c);
    h[idx] = hh;
    hf16[idx] = (f16)hh;
    out_t[idx] = hh;
}

// write val row into memory value store (pos = t*32 while < N, else argmax pos)
__global__ void mem_write_k(const float* __restrict__ valbuf, float* __restrict__ valmem,
                            const int* __restrict__ pos, int t) {
    int b = blockIdx.x, j = threadIdx.x;   // 128 threads
    int ts = t * LSTEPS;
    int p  = (ts < NSLOT) ? ts : pos[b];
    valmem[((size_t)b * NSLOT + p) * CDIM + j] = valbuf[(size_t)b * CDIM + j];
}

// ---------------------------------------------------------------------------
// Host side
// ---------------------------------------------------------------------------
static inline void gemm(const f16* A, const f16* Bm, const float* Add, float* Cp,
                        int M, int Nn, int K, hipStream_t s) {
    dim3 g(M / 64, Nn / 64);
    if (Add) gemm_f16_wmma<true ><<<g, 256, 0, s>>>(A, Bm, Add, Cp, M, Nn, K);
    else     gemm_f16_wmma<false><<<g, 256, 0, s>>>(A, Bm, nullptr, Cp, M, Nn, K);
}

extern "C" void kernel_launch(void* const* d_in, const int* in_sizes, int n_in,
                              void* d_out, int out_size, void* d_ws, size_t ws_size,
                              hipStream_t stream) {
    const float* inp      = (const float*)d_in[0];
    const float* hid      = (const float*)d_in[1];
    const float* mem_bias = (const float*)d_in[2];
    const float* W_h2w    = (const float*)d_in[3];
    const float* W_i2w    = (const float*)d_in[4];
    const float* W_m2w    = (const float*)d_in[5];
    const float* W_u2w    = (const float*)d_in[6];
    const float* W_h2g    = (const float*)d_in[7];
    const float* W_i2g    = (const float*)d_in[8];
    const float* W_r2g    = (const float*)d_in[9];
    const float* W_h2ab   = (const float*)d_in[10];
    const float* W_i2ab   = (const float*)d_in[11];
    const float* W_r2ab   = (const float*)d_in[12];
    const float* W_h2c    = (const float*)d_in[13];
    const float* W_i2c    = (const float*)d_in[14];
    const float* W_r2c    = (const float*)d_in[15];
    const float* atten    = (const float*)d_in[16];
    // d_in[17] W_h2tau, d_in[18] b_h2tau: tau only rescales softmax; the ST
    // forward value is the argmax one-hot, which is tau-invariant -> unused.
    const float* W_h2m    = (const float*)d_in[19];
    float* out = (float*)d_out;
    (void)in_sizes; (void)n_in; (void)out_size; (void)ws_size;

    char* p = (char*)d_ws;
    auto alloc = [&](size_t bytes) -> char* {
        char* r = p; p += (bytes + 255) & ~(size_t)255; return r;
    };
    // f16 buffers
    f16* inp_h   = (f16*)alloc(sizeof(f16) * (size_t)LB * IDIM);
    f16* wh2w_h  = (f16*)alloc(sizeof(f16) * HDIM * ACD);
    f16* wi2w_h  = (f16*)alloc(sizeof(f16) * IDIM * ACD);
    f16* wu2w_h  = (f16*)alloc(sizeof(f16) * NSLOT * ACD);
    f16* wm2wA_h = (f16*)alloc(sizeof(f16) * ADIM * ACD);
    f16* wm2wC_h = (f16*)alloc(sizeof(f16) * CDIM * ACD);
    f16* wh2c_h  = (f16*)alloc(sizeof(f16) * HDIM * HDIM);
    f16* wi2c_h  = (f16*)alloc(sizeof(f16) * IDIM * HDIM);
    f16* wr2c_h  = (f16*)alloc(sizeof(f16) * ACD * HDIM);
    f16* wh2m_h  = (f16*)alloc(sizeof(f16) * HDIM * CDIM);
    f16* addr_h  = (f16*)alloc(sizeof(f16) * NSLOT * ADIM);
    f16* hf16    = (f16*)alloc(sizeof(f16) * BATCH * HDIM);
    f16* uh      = (f16*)alloc(sizeof(f16) * BATCH * NSLOT);
    f16* rf16    = (f16*)alloc(sizeof(f16) * BATCH * ACD);
    // f32 buffers
    float* addr_part = (float*)alloc(sizeof(float) * NSLOT * ACD);
    float* embW   = (float*)alloc(sizeof(float) * (size_t)LB * ACD);
    float* embC   = (float*)alloc(sizeof(float) * (size_t)LB * HDIM);
    float* embG   = (float*)alloc(sizeof(float) * LB * 3);
    float* embAB  = (float*)alloc(sizeof(float) * LB * 2);
    float* h      = (float*)alloc(sizeof(float) * BATCH * HDIM);
    float* cc     = (float*)alloc(sizeof(float) * BATCH * HDIM);
    float* w_sum  = (float*)alloc(sizeof(float) * BATCH * NSLOT);
    float* base   = (float*)alloc(sizeof(float) * BATCH * ACD);
    float* logits = (float*)alloc(sizeof(float) * BATCH * NSLOT);
    float* rbuf   = (float*)alloc(sizeof(float) * BATCH * ACD);
    float* fio    = (float*)alloc(sizeof(float) * BATCH * 3);
    float* albe   = (float*)alloc(sizeof(float) * BATCH * 2);
    float* P1     = (float*)alloc(sizeof(float) * BATCH * HDIM);
    float* P2     = (float*)alloc(sizeof(float) * BATCH * HDIM);
    float* valbuf = (float*)alloc(sizeof(float) * BATCH * CDIM);
    float* valmem = (float*)alloc(sizeof(float) * (size_t)BATCH * NSLOT * CDIM);
    int*   pos    = (int*)alloc(sizeof(int) * BATCH);

    auto cvt = [&](const float* s, f16* d, int n) {
        f32_to_f16_k<<<(n + 255) / 256, 256, 0, stream>>>(s, d, n);
    };
    auto zero = [&](float* q, int n) {
        zero_f32_k<<<(n + 255) / 256, 256, 0, stream>>>(q, n);
    };

    // ---- one-time prep --------------------------------------------------
    cvt(inp,   inp_h,  LB * IDIM);
    cvt(W_h2w, wh2w_h, HDIM * ACD);
    cvt(W_i2w, wi2w_h, IDIM * ACD);
    cvt(W_u2w, wu2w_h, NSLOT * ACD);
    cvt(W_m2w,              wm2wA_h, ADIM * ACD);           // rows 0..A-1
    cvt(W_m2w + ADIM * ACD, wm2wC_h, CDIM * ACD);           // rows A..AC-1
    cvt(W_h2c, wh2c_h, HDIM * HDIM);
    cvt(W_i2c, wi2c_h, IDIM * HDIM);
    cvt(W_r2c, wr2c_h, ACD * HDIM);
    cvt(W_h2m, wh2m_h, HDIM * CDIM);
    extract_addr_k<<<(NSLOT * ADIM + 255) / 256, 256, 0, stream>>>(mem_bias, addr_h);
    init_h_k<<<(BATCH * HDIM + 255) / 256, 256, 0, stream>>>(hid, h, hf16, BATCH * HDIM);
    zero(cc, BATCH * HDIM);
    zero(w_sum, BATCH * NSLOT);
    zero(valmem, BATCH * NSLOT * CDIM);

    // constant addr contribution of mem @ W_m2w
    gemm(addr_h, wm2wA_h, nullptr, addr_part, NSLOT, ACD, ADIM, stream);
    // hoisted emb projections for all timesteps
    gemm(inp_h, wi2w_h, nullptr, embW, LB, ACD, IDIM, stream);
    gemm(inp_h, wi2c_h, nullptr, embC, LB, HDIM, IDIM, stream);
    emb_small_k<<<(LB * 5 + 255) / 256, 256, 0, stream>>>(inp, W_i2g, W_i2ab, embG, embAB);

    // ---- recurrent steps ------------------------------------------------
    for (int t = 0; t < LSTEPS; ++t) {
        unorm_k<<<BATCH, 128, 0, stream>>>(w_sum, uh);
        // base = h@W_h2w + emb_w[t] + u@W_u2w
        gemm(hf16, wh2w_h, embW + (size_t)t * BATCH * ACD, base, BATCH, ACD, HDIM, stream);
        gemm(uh, wu2w_h, base, base, BATCH, ACD, NSLOT, stream);
        // fused: valprod GEMM + tanh + atten dot -> logits
        logits_kernel<<<dim3(BATCH, NSLOT / 64), 256, 0, stream>>>(
            valmem, wm2wC_h, base, addr_part, atten, logits);
        argmax_pick_k<<<BATCH, 128, 0, stream>>>(logits, w_sum, pos, mem_bias, valmem, rbuf, rf16);
        gates_k<<<BATCH, 256, 0, stream>>>(h, rbuf, W_h2g, W_r2g, W_h2ab, W_r2ab,
                                           embG + (size_t)t * BATCH * 3,
                                           embAB + (size_t)t * BATCH * 2, fio, albe);
        gemm(hf16, wh2c_h, nullptr, P1, BATCH, HDIM, HDIM, stream);
        gemm(rf16, wr2c_h, nullptr, P2, BATCH, HDIM, ACD, stream);
        state_update_k<<<(BATCH * HDIM) / 256, 256, 0, stream>>>(
            P1, P2, embC + (size_t)t * BATCH * HDIM, fio, albe, cc, h, hf16,
            out + (size_t)t * BATCH * HDIM);
        // val = h_new @ W_h2m, written into slot (t*32 < N ? t*32 : argmax)
        gemm(hf16, wh2m_h, nullptr, valbuf, BATCH, CDIM, HDIM, stream);
        mem_write_k<<<BATCH, 128, 0, stream>>>(valbuf, valmem, pos, t);
    }
}
